// QuaildFacilityLocationLoss_7224134992492
// MI455X (gfx1250) — compile-verified, matrix operand order checked by reading, not run on hardware
//
#include <hip/hip_runtime.h>

// ---------------------------------------------------------------------------
// loss = sum_i max_j (A·Bᵀ)[i][j] + sum_j max_i (A·Bᵀ)[i][j]
// N=M=16384, K=1024, f32 inputs.
//
// Fast path (needs ~128.1 MB of d_ws):
//   1) fl_convert: one-shot f32 -> bf16 hi/lo split arrays (each element
//      converted once, not 128x).
//   2) fl_gemm_max_pre: 128x128 tiles, double-buffered LDS staged with
//      gfx1250 async direct-to-LDS loads (ASYNCcnt), 3-product bf16 split
//      WMMA (~fp32 accuracy), on-the-fly row/col max -> ordered-uint atomics.
//   3) fl_reduce: decode keys, sum, atomicAdd scalar.
// Fallback path (small d_ws): round-1 inline-convert GEMM.
// ---------------------------------------------------------------------------

typedef __attribute__((ext_vector_type(16))) __bf16   v16bf;
typedef __attribute__((ext_vector_type(8)))  float    v8f;
typedef __attribute__((ext_vector_type(4)))  unsigned v4u;
typedef __attribute__((ext_vector_type(4)))  int      v4i;

#define N_ROWS 16384
#define K_DIM  1024
#define TILE   128
#define BK     32
#define LDST   40      // padded LDS row stride in bf16 elems (80B -> conflict-free b128)

// ---- order-preserving f32 <-> u32 keys ----
__device__ __forceinline__ unsigned enc_f32(float f) {
  unsigned u = __float_as_uint(f);
  return (u & 0x80000000u) ? ~u : (u | 0x80000000u);
}
__device__ __forceinline__ float dec_f32(unsigned k) {
  unsigned u = (k & 0x80000000u) ? (k & 0x7FFFFFFFu) : ~k;
  return __uint_as_float(u);
}

// ---- f32 <-> bf16 ----
__device__ __forceinline__ unsigned short bf16_rn(float f) {
  unsigned u = __float_as_uint(f);
  u += 0x7FFFu + ((u >> 16) & 1u);
  return (unsigned short)(u >> 16);
}
__device__ __forceinline__ float bf16_f32(unsigned short h) {
  return __uint_as_float(((unsigned)h) << 16);
}

// ---- async direct-to-LDS (gfx1250) with sync fallback ----
#if defined(__has_builtin)
#if __has_builtin(__builtin_amdgcn_global_load_async_to_lds_b128)
#define HAVE_ASYNC_LDS 1
#endif
#endif
#ifndef HAVE_ASYNC_LDS
#define HAVE_ASYNC_LDS 0
#endif

__device__ __forceinline__ void copy16_to_lds(const unsigned short* gsrc,
                                              unsigned short* ldst) {
#if HAVE_ASYNC_LDS
  // builtin signature (from probe diagnostic): param0 = v4i __device__* (AS1),
  // param1 = LDS pointer, then imm offset + cpol.
  __builtin_amdgcn_global_load_async_to_lds_b128(
      (__attribute__((address_space(1))) v4i*)gsrc,
      (__attribute__((address_space(3))) v4i*)ldst, 0, 0);
#else
  *(uint4*)ldst = *(const uint4*)gsrc;
#endif
}
__device__ __forceinline__ void wait_async_copies() {
#if HAVE_ASYNC_LDS
  asm volatile("s_wait_asynccnt 0x0" ::: "memory");
#endif
}

// ---- WMMA 16-bit operand fragment from LDS tile ----
union Frag { v16bf v; struct { v4u lo; v4u hi; } q; };

__device__ __forceinline__ v16bf load_frag(const unsigned short* tile, int row, int khalf) {
  Frag f;
  const unsigned short* p = tile + row * LDST + khalf;
  f.q.lo = *(const v4u*)(p);        // K = khalf .. khalf+7
  f.q.hi = *(const v4u*)(p + 16);   // K = khalf+16 .. khalf+23
  return f.v;
}

// ---------------------------------------------------------------------------
__global__ void fl_init(unsigned* __restrict__ umax, float* __restrict__ out) {
  int i = blockIdx.x * blockDim.x + threadIdx.x;
  if (i < 2 * N_ROWS) umax[i] = 0u;
  if (i == 0) out[0] = 0.0f;
}

// ---------------------------------------------------------------------------
// One-shot f32 -> (bf16 hi, bf16 lo) split. n4 = element count / 4.
__global__ __launch_bounds__(256)
void fl_convert(const float* __restrict__ src, unsigned short* __restrict__ hi,
                unsigned short* __restrict__ lo, int n4) {
  for (int i = blockIdx.x * blockDim.x + threadIdx.x; i < n4;
       i += gridDim.x * blockDim.x) {
    float4 v = ((const float4*)src)[i];
    const float f[4] = {v.x, v.y, v.z, v.w};
    unsigned short h[4], l[4];
#pragma unroll
    for (int e = 0; e < 4; ++e) {
      h[e] = bf16_rn(f[e]);
      l[e] = bf16_rn(f[e] - bf16_f32(h[e]));
    }
    uint2 hp, lp;
    hp.x = (unsigned)h[0] | ((unsigned)h[1] << 16);
    hp.y = (unsigned)h[2] | ((unsigned)h[3] << 16);
    lp.x = (unsigned)l[0] | ((unsigned)l[1] << 16);
    lp.y = (unsigned)l[2] | ((unsigned)l[3] << 16);
    ((uint2*)hi)[i] = hp;
    ((uint2*)lo)[i] = lp;
  }
}

// ---------------------------------------------------------------------------
// Shared epilogue: per-wave row/col max -> LDS keys -> global keys.
__device__ __forceinline__ void max_epilogue(const v8f acc[4][2], int t, int lane,
                                             int wm, int wn, int lrow,
                                             unsigned* rowmax_s, unsigned* colmax_s,
                                             unsigned* umax_a, unsigned* umax_b,
                                             long rowBase, long colBase) {
#pragma unroll
  for (int i = 0; i < 4; ++i) {
    float rm[8];
#pragma unroll
    for (int v = 0; v < 8; ++v) rm[v] = fmaxf(acc[i][0][v], acc[i][1][v]);
#pragma unroll
    for (int off = 8; off >= 1; off >>= 1) {
#pragma unroll
      for (int v = 0; v < 8; ++v) rm[v] = fmaxf(rm[v], __shfl_xor(rm[v], off, 32));
    }
    if (lrow == 0) {
      int rbase = wm * 64 + i * 16 + ((lane >> 4) << 3);
#pragma unroll
      for (int v = 0; v < 8; ++v) atomicMax(&rowmax_s[rbase + v], enc_f32(rm[v]));
    }
  }
#pragma unroll
  for (int j = 0; j < 2; ++j) {
    float cm = -3.0e38f;
#pragma unroll
    for (int i = 0; i < 4; ++i) {
#pragma unroll
      for (int v = 0; v < 8; ++v) cm = fmaxf(cm, acc[i][j][v]);
    }
    cm = fmaxf(cm, __shfl_xor(cm, 16, 32));
    if (lane < 16) atomicMax(&colmax_s[wn * 32 + j * 16 + lane], enc_f32(cm));
  }
  __syncthreads();
  if (t < TILE) {
    atomicMax(&umax_a[rowBase + t], rowmax_s[t]);
    atomicMax(&umax_b[colBase + t], colmax_s[t]);
  }
}

// ---------------------------------------------------------------------------
// FAST PATH: precomputed bf16 tiles, async-to-LDS double-buffered pipeline.
__global__ __launch_bounds__(256)
void fl_gemm_max_pre(const unsigned short* __restrict__ Ahi,
                     const unsigned short* __restrict__ Alo,
                     const unsigned short* __restrict__ Bhi,
                     const unsigned short* __restrict__ Blo,
                     unsigned* __restrict__ umax_a, unsigned* __restrict__ umax_b) {
  __shared__ unsigned short sT[2][4][TILE * LDST];  // [buf][Ahi,Alo,Bhi,Blo]
  __shared__ unsigned rowmax_s[TILE];
  __shared__ unsigned colmax_s[TILE];

  const int t     = threadIdx.x;
  const int lane  = t & 31;
  const int wave  = t >> 5;
  const int wm    = wave & 1;
  const int wn    = wave >> 1;
  const int lrow  = lane & 15;
  const int khalf = (lane >> 4) << 3;

  const long rowBase = (long)blockIdx.y * TILE;
  const long colBase = (long)blockIdx.x * TILE;

  if (t < TILE) { rowmax_s[t] = 0u; colmax_s[t] = 0u; }

  // Stage one k-step: 4 tiles of 128 rows x 32 bf16 (512 x 16B chunks each).
  auto issue = [&](int k0, int buf) {
#pragma unroll
    for (int i = 0; i < 2; ++i) {
      int c = t + (i << 8);          // 0..511
      int r = c >> 2;                // tile row
      int q = c & 3;                 // 16B chunk within 64B row
      long aoff = (rowBase + r) * K_DIM + k0 + (q << 3);
      long boff = (colBase + r) * K_DIM + k0 + (q << 3);
      int  loff = r * LDST + (q << 3);
      copy16_to_lds(Ahi + aoff, &sT[buf][0][loff]);
      copy16_to_lds(Alo + aoff, &sT[buf][1][loff]);
      copy16_to_lds(Bhi + boff, &sT[buf][2][loff]);
      copy16_to_lds(Blo + boff, &sT[buf][3][loff]);
    }
  };

  v8f acc[4][2] = {};

  issue(0, 0);
  wait_async_copies();
  __syncthreads();

  int p = 0;
  for (int k0 = 0; k0 < K_DIM; k0 += BK) {
    if (k0 + BK < K_DIM) issue(k0 + BK, p ^ 1);  // overlap next stage with WMMA

    v16bf bhiF[2], bloF[2];
#pragma unroll
    for (int j = 0; j < 2; ++j) {
      int r = wn * 32 + j * 16 + lrow;
      bhiF[j] = load_frag(sT[p][2], r, khalf);
      bloF[j] = load_frag(sT[p][3], r, khalf);
    }
#pragma unroll
    for (int i = 0; i < 4; ++i) {
      int r = wm * 64 + i * 16 + lrow;
      v16bf ahi = load_frag(sT[p][0], r, khalf);
      v16bf alo = load_frag(sT[p][1], r, khalf);
#pragma unroll
      for (int j = 0; j < 2; ++j) {
        acc[i][j] = __builtin_amdgcn_wmma_f32_16x16x32_bf16(
            false, ahi, false, bhiF[j], (short)0, acc[i][j], false, false);
        acc[i][j] = __builtin_amdgcn_wmma_f32_16x16x32_bf16(
            false, ahi, false, bloF[j], (short)0, acc[i][j], false, false);
        acc[i][j] = __builtin_amdgcn_wmma_f32_16x16x32_bf16(
            false, alo, false, bhiF[j], (short)0, acc[i][j], false, false);
      }
    }

    wait_async_copies();
    __syncthreads();
    p ^= 1;
  }

  max_epilogue(acc, t, lane, wm, wn, lrow, rowmax_s, colmax_s,
               umax_a, umax_b, rowBase, colBase);
}

// ---------------------------------------------------------------------------
// FALLBACK: inline f32->bf16 split per tile (round-1 kernel).
__global__ __launch_bounds__(256)
void fl_gemm_max_inline(const float* __restrict__ A, const float* __restrict__ B,
                        unsigned* __restrict__ umax_a, unsigned* __restrict__ umax_b) {
  __shared__ unsigned short sAhi[TILE * LDST];
  __shared__ unsigned short sAlo[TILE * LDST];
  __shared__ unsigned short sBhi[TILE * LDST];
  __shared__ unsigned short sBlo[TILE * LDST];
  __shared__ unsigned rowmax_s[TILE];
  __shared__ unsigned colmax_s[TILE];

  const int t     = threadIdx.x;
  const int lane  = t & 31;
  const int wave  = t >> 5;
  const int wm    = wave & 1;
  const int wn    = wave >> 1;
  const int lrow  = lane & 15;
  const int khalf = (lane >> 4) << 3;

  const long rowBase = (long)blockIdx.y * TILE;
  const long colBase = (long)blockIdx.x * TILE;

  if (t < TILE) { rowmax_s[t] = 0u; colmax_s[t] = 0u; }

  v8f acc[4][2] = {};

  for (int k0 = 0; k0 < K_DIM; k0 += BK) {
    __syncthreads();
#pragma unroll
    for (int i = 0; i < 4; ++i) {
      int c = t + (i << 8);
      int r = c >> 3;
      int q = c & 7;
      const float4 av = *(const float4*)(A + (rowBase + r) * K_DIM + k0 + (q << 2));
      const float4 bv = *(const float4*)(B + (colBase + r) * K_DIM + k0 + (q << 2));
      const float a4[4] = {av.x, av.y, av.z, av.w};
      const float b4[4] = {bv.x, bv.y, bv.z, bv.w};
      int base = r * LDST + (q << 2);
#pragma unroll
      for (int e = 0; e < 4; ++e) {
        unsigned short ah = bf16_rn(a4[e]);
        sAhi[base + e] = ah;
        sAlo[base + e] = bf16_rn(a4[e] - bf16_f32(ah));
        unsigned short bh = bf16_rn(b4[e]);
        sBhi[base + e] = bh;
        sBlo[base + e] = bf16_rn(b4[e] - bf16_f32(bh));
      }
    }
    __syncthreads();

    v16bf bhiF[2], bloF[2];
#pragma unroll
    for (int j = 0; j < 2; ++j) {
      int r = wn * 32 + j * 16 + lrow;
      bhiF[j] = load_frag(sBhi, r, khalf);
      bloF[j] = load_frag(sBlo, r, khalf);
    }
#pragma unroll
    for (int i = 0; i < 4; ++i) {
      int r = wm * 64 + i * 16 + lrow;
      v16bf ahi = load_frag(sAhi, r, khalf);
      v16bf alo = load_frag(sAlo, r, khalf);
#pragma unroll
      for (int j = 0; j < 2; ++j) {
        acc[i][j] = __builtin_amdgcn_wmma_f32_16x16x32_bf16(
            false, ahi, false, bhiF[j], (short)0, acc[i][j], false, false);
        acc[i][j] = __builtin_amdgcn_wmma_f32_16x16x32_bf16(
            false, ahi, false, bloF[j], (short)0, acc[i][j], false, false);
        acc[i][j] = __builtin_amdgcn_wmma_f32_16x16x32_bf16(
            false, alo, false, bhiF[j], (short)0, acc[i][j], false, false);
      }
    }
  }

  max_epilogue(acc, t, lane, wm, wn, lrow, rowmax_s, colmax_s,
               umax_a, umax_b, rowBase, colBase);
}

// ---------------------------------------------------------------------------
__global__ __launch_bounds__(256)
void fl_reduce(const unsigned* __restrict__ umax, float* __restrict__ out) {
  float s = 0.0f;
  const int total = 2 * N_ROWS;  // LAMBD == 1.0
  for (int i = blockIdx.x * blockDim.x + threadIdx.x; i < total;
       i += gridDim.x * blockDim.x)
    s += dec_f32(umax[i]);
#pragma unroll
  for (int off = 16; off >= 1; off >>= 1) s += __shfl_xor(s, off, 32);
  __shared__ float ws[8];
  int lane = threadIdx.x & 31, wave = threadIdx.x >> 5;
  if (lane == 0) ws[wave] = s;
  __syncthreads();
  if (threadIdx.x == 0) {
    float tot = 0.0f;
#pragma unroll
    for (int w = 0; w < 8; ++w) tot += ws[w];
    atomicAdd(out, tot);
  }
}

// ---------------------------------------------------------------------------
extern "C" void kernel_launch(void* const* d_in, const int* in_sizes, int n_in,
                              void* d_out, int out_size, void* d_ws, size_t ws_size,
                              hipStream_t stream) {
  const float* A = (const float*)d_in[0];
  const float* B = (const float*)d_in[1];
  float* out = (float*)d_out;
  unsigned* umax = (unsigned*)d_ws;

  const size_t UMAX_BYTES = (size_t)2 * N_ROWS * sizeof(unsigned);
  const size_t NK         = (size_t)N_ROWS * K_DIM;
  const size_t ARR_BYTES  = NK * sizeof(unsigned short);

  fl_init<<<(2 * N_ROWS + 255) / 256, 256, 0, stream>>>(umax, out);

  dim3 grid(N_ROWS / TILE, N_ROWS / TILE);

  if (ws_size >= UMAX_BYTES + 4 * ARR_BYTES) {
    unsigned short* base = (unsigned short*)((char*)d_ws + UMAX_BYTES);
    unsigned short* Ahi = base;
    unsigned short* Alo = base + NK;
    unsigned short* Bhi = base + 2 * NK;
    unsigned short* Blo = base + 3 * NK;
    fl_convert<<<4096, 256, 0, stream>>>(A, Ahi, Alo, (int)(NK / 4));
    fl_convert<<<4096, 256, 0, stream>>>(B, Bhi, Blo, (int)(NK / 4));
    fl_gemm_max_pre<<<grid, 256, 0, stream>>>(Ahi, Alo, Bhi, Blo, umax, umax + N_ROWS);
  } else {
    fl_gemm_max_inline<<<grid, 256, 0, stream>>>(A, B, umax, umax + N_ROWS);
  }

  fl_reduce<<<64, 256, 0, stream>>>(umax, out);
}
